// Attention_33200097198117
// MI455X (gfx1250) — matrix operand order repaired and checked by reading, compiled, not run on hardware
//
#include <hip/hip_runtime.h>

typedef __bf16 bf16_t;
typedef bf16_t v16bf __attribute__((ext_vector_type(16)));
typedef float  v8f   __attribute__((ext_vector_type(8)));
typedef int    v4i_t __attribute__((vector_size(16)));   // matches builtin's V4i prototype

#define NUMH 12
#define BATCH 16
#define SEQ 1025
#define CH 768
#define DH 64
#define QKVN (3*CH)

#define AS1 __attribute__((address_space(1)))
#define AS3 __attribute__((address_space(3)))

#if defined(__has_builtin)
# if __has_builtin(__builtin_amdgcn_global_load_async_to_lds_b128) && __has_builtin(__builtin_amdgcn_s_wait_asynccnt)
#  define HAVE_ASYNC_LDS 1
# endif
#endif

#ifdef HAVE_ASYNC_LDS
// prototype (from hipcc diagnostic): (v4i AS1*, v4i AS3*, imm offset, imm cpol)
# define CP_B128(ldsDst, gSrc) \
    __builtin_amdgcn_global_load_async_to_lds_b128( \
        (AS1 v4i_t*)(v4i_t*)const_cast<void*>((const void*)(gSrc)), \
        (AS3 v4i_t*)(v4i_t*)(void*)(ldsDst), 0, 0)
# define ASYNC_WAIT(n) __builtin_amdgcn_s_wait_asynccnt(n)
#else
# define CP_B128(ldsDst, gSrc) (*(uint4*)(ldsDst) = *(const uint4*)(gSrc))
# define ASYNC_WAIT(n) ((void)0)
#endif

__device__ __forceinline__ v8f wmma_bf(v16bf a, v16bf b, v8f c) {
  return __builtin_amdgcn_wmma_f32_16x16x32_bf16(false, a, false, b, (short)0, c, false, false);
}

union FragU { v16bf v; uint4 q[2]; };

// A-frag 16x32 (row-major source, stride ld): element j -> K = 16*(j/8) + 8*half + (j%8)
__device__ __forceinline__ v16bf load_a16x32(const bf16_t* base, int ld, int lane) {
  int m = lane & 15, hf = (lane >> 4) & 1;
  const bf16_t* p = base + m * ld;
  FragU u;
  u.q[0] = *(const uint4*)(p + 8*hf);
  u.q[1] = *(const uint4*)(p + 16 + 8*hf);
  return u.v;
}

// B-frag 32x16 from TRANSPOSED [N][K] tile (stride ld): element j -> K = 16*(j/8) + 8*half + (j%8)
__device__ __forceinline__ v16bf load_b32x16(const bf16_t* base, int ld, int lane) {
  int n = lane & 15, hf = (lane >> 4) & 1;
  const bf16_t* p = base + n * ld;
  FragU u;
  u.q[0] = *(const uint4*)(p + 8*hf);
  u.q[1] = *(const uint4*)(p + 16 + 8*hf);
  return u.v;
}

__global__ void f32_to_bf16_kernel(const float* __restrict__ in, bf16_t* __restrict__ out, int n) {
  int i = blockIdx.x * blockDim.x + threadIdx.x;
  if (i < n) out[i] = (bf16_t)in[i];
}

// Fused convert + transpose: WT[n*K + k] = bf16(W[k*N + n]); 32x32 LDS tiles.
__global__ __launch_bounds__(256)
void transpose_w_kernel(const float* __restrict__ W, bf16_t* __restrict__ WT, int K, int N) {
  __shared__ float tile[32][33];
  int k0 = blockIdx.x * 32, n0 = blockIdx.y * 32;
  int tx = threadIdx.x & 31, ty = threadIdx.x >> 5;  // 8 row-groups
  #pragma unroll
  for (int i = ty; i < 32; i += 8) tile[i][tx] = W[(size_t)(k0 + i) * N + n0 + tx];
  __syncthreads();
  #pragma unroll
  for (int i = ty; i < 32; i += 8) WT[(size_t)(n0 + i) * K + k0 + tx] = (bf16_t)tile[tx][i];
}

// -------------------- GEMM core: (16400 x 768) @ WT(NW x 768)^T, double-buffered async LDS
// MODE 0: scatter bf16 into (3,B,H,N,Dh).  MODE 1: f32 + bias into (16400, NW).
template <int NW, int MODE>
__global__ __launch_bounds__(256)
void gemm_kernel(const bf16_t* __restrict__ A, const bf16_t* __restrict__ WT,
                 bf16_t* __restrict__ outb, float* __restrict__ outf,
                 const float* __restrict__ bias) {
  const int M = BATCH * SEQ, K = CH;
  __shared__ __align__(16) bf16_t sA[2][128 * 40];
  __shared__ __align__(16) bf16_t sB[2][64 * 40];
  int t = threadIdx.x, lane = t & 31, wave = t >> 5;
  int wm = wave & 3, wn = wave >> 2;
  int m0 = blockIdx.x * 128, n0 = blockIdx.y * 64;
  int nl = lane & 15, hf = (lane >> 4) & 1;

  // each thread issues exactly 3 in-order b128 copies per stage (2 for A, 1 for B)
  auto stage = [&](int buf, int k0) {
    int row = t >> 1, cc = (t & 1) * 16;
    int gr = m0 + row; if (gr >= M) gr = M - 1;
    const bf16_t* srcA = A + (size_t)gr * K + k0 + cc;
    CP_B128(&sA[buf][row * 40 + cc],     srcA);
    CP_B128(&sA[buf][row * 40 + cc + 8], srcA + 8);
    int nn = t >> 2, c = (t & 3) * 8;
    const bf16_t* srcB = WT + (size_t)(n0 + nn) * K + k0 + c;
    CP_B128(&sB[buf][nn * 40 + c], srcB);
    __builtin_prefetch(srcA + 32, 0, 1);
  };

  const v8f vzero = {0.f,0.f,0.f,0.f,0.f,0.f,0.f,0.f};
  v8f acc[2][2] = {{vzero, vzero}, {vzero, vzero}};

  stage(0, 0);
  int cur = 0;
  for (int k0 = 0; k0 < K; k0 += 32) {
    __syncthreads();                       // prev buffer fully consumed by all waves
    if (k0 + 32 < K) { stage(cur ^ 1, k0 + 32); ASYNC_WAIT(3); }
    else             { ASYNC_WAIT(0); }
    __syncthreads();                       // current buffer staged by all waves
    v16bf a0 = load_a16x32(&sA[cur][(32*wm)      * 40], 40, lane);
    v16bf a1 = load_a16x32(&sA[cur][(32*wm + 16) * 40], 40, lane);
    v16bf b0 = load_b32x16(&sB[cur][(32*wn)      * 40], 40, lane);
    v16bf b1 = load_b32x16(&sB[cur][(32*wn + 16) * 40], 40, lane);
    acc[0][0] = wmma_bf(a0, b0, acc[0][0]);
    acc[0][1] = wmma_bf(a0, b1, acc[0][1]);
    acc[1][0] = wmma_bf(a1, b0, acc[1][0]);
    acc[1][1] = wmma_bf(a1, b1, acc[1][1]);
    cur ^= 1;
  }
  #pragma unroll
  for (int ts = 0; ts < 2; ++ts)
    #pragma unroll
    for (int ns = 0; ns < 2; ++ns)
      #pragma unroll
      for (int r = 0; r < 8; ++r) {
        int row = m0 + 32*wm + 16*ts + r + 8*hf;
        int col = n0 + 32*wn + 16*ns + nl;
        if (row < M) {
          if constexpr (MODE == 0) {
            int b = row / SEQ, n = row % SEQ;
            int j3 = col / CH, rem = col % CH, h = rem / DH, d = rem % DH;
            size_t off = ((((size_t)j3 * BATCH + b) * NUMH + h) * SEQ + n) * DH + d;
            outb[off] = (bf16_t)acc[ts][ns][r];
          } else {
            outf[(size_t)row * NW + col] = acc[ts][ns][r] + bias[col];
          }
        }
      }
}

// -------------------- RoPE on Q,K patch tokens (in place, per (d, d+32) pair)
__global__ void rope_kernel(bf16_t* __restrict__ qkv,
                            const float* __restrict__ sinp, const float* __restrict__ cosp) {
  int i = blockIdx.x * blockDim.x + threadIdx.x;   // 2*16*12*1024*32 threads
  int d = i & 31;
  int r = i >> 5;
  int n1 = r & 1023; r >>= 10;
  int h = r % NUMH;  r /= NUMH;
  int b = r & 15;
  int j3 = r >> 4;                                  // 0 = q, 1 = k
  bf16_t* p = qkv + ((((size_t)j3 * BATCH + b) * NUMH + h) * SEQ + (n1 + 1)) * DH;
  float v0 = (float)p[d], v1 = (float)p[d + 32];
  float s0 = sinp[n1 * DH + d],      c0 = cosp[n1 * DH + d];
  float s1 = sinp[n1 * DH + d + 32], c1 = cosp[n1 * DH + d + 32];
  p[d]      = (bf16_t)(v0 * c0 - v1 * s0);
  p[d + 32] = (bf16_t)(v1 * c1 + v0 * s1);
}

// -------------------- Flash attention: one block per (b*h, 128-row q block)
__global__ __launch_bounds__(256)
void flash_attn_kernel(const bf16_t* __restrict__ qb, const bf16_t* __restrict__ kb,
                       const bf16_t* __restrict__ vb, bf16_t* __restrict__ out) {
  __shared__ __align__(16) bf16_t sK[32 * 72];       // [key][d], matches B-frag layout
  __shared__ __align__(16) bf16_t sV[64 * 40];       // [d][key], transposed
  __shared__ __align__(16) float  sS[8][16 * 32];    // per-wave score scratch
  __shared__ __align__(16) bf16_t sP[8][16 * 32];    // per-wave P (A-frag layout)
  __shared__ float sM[8][16], sL[8][16], sF[8][16];

  const float scale = 0.125f;  // 64^-0.5
  int t = threadIdx.x, lane = t & 31, wv = t >> 5;
  int bh = blockIdx.x;
  int b = bh / NUMH, h = bh % NUMH;
  int q0 = blockIdx.y * 128 + wv * 16;
  int nl = lane & 15, hf = (lane >> 4) & 1;

  const bf16_t* qbase = qb + (size_t)bh * SEQ * DH;
  const bf16_t* kbase = kb + (size_t)bh * SEQ * DH;
  const bf16_t* vbase = vb + (size_t)bh * SEQ * DH;

  v16bf qf[2];
  {
    int qr = q0 + nl; if (qr >= SEQ) qr = SEQ - 1;
    const bf16_t* p = qbase + (size_t)qr * DH;
    #pragma unroll
    for (int dk = 0; dk < 2; ++dk) {
      FragU u;
      u.q[0] = *(const uint4*)(p + 32*dk + 8*hf);
      u.q[1] = *(const uint4*)(p + 32*dk + 16 + 8*hf);
      qf[dk] = u.v;
    }
  }

  const v8f vzero = {0.f,0.f,0.f,0.f,0.f,0.f,0.f,0.f};
  v8f accO[4] = {vzero, vzero, vzero, vzero};
  if (lane < 16) { sM[wv][lane] = -1e30f; sL[wv][lane] = 0.f; }

  const int nkb = (SEQ + 31) / 32;  // 33
  for (int kbk = 0; kbk < nkb; ++kbk) {
    __syncthreads();
    { // stage K (async direct) and V (transposed via VGPRs), zero-fill OOB keys
      int i = t >> 3, c = t & 7;
      int key = kbk * 32 + i;
      if (key < SEQ) {
        CP_B128(&sK[i * 72 + c * 8], kbase + (size_t)key * DH + c * 8);
        uint4 v4 = *(const uint4*)(vbase + (size_t)key * DH + c * 8);
        bf16_t vt[8]; *(uint4*)vt = v4;
        #pragma unroll
        for (int e = 0; e < 8; ++e) sV[(c * 8 + e) * 40 + i] = vt[e];
      } else {
        uint4 z = {0,0,0,0};
        *(uint4*)&sK[i * 72 + c * 8] = z;
        #pragma unroll
        for (int e = 0; e < 8; ++e) sV[(c * 8 + e) * 40 + i] = (bf16_t)0.f;
      }
    }
    ASYNC_WAIT(0);
    __syncthreads();
    // S = Q K^T  (16 q-rows x 32 keys per wave)
    #pragma unroll
    for (int kn = 0; kn < 2; ++kn) {
      v8f s = vzero;
      s = wmma_bf(qf[0], load_b32x16(sK + 16*kn*72 + 0,  72, lane), s);
      s = wmma_bf(qf[1], load_b32x16(sK + 16*kn*72 + 32, 72, lane), s);
      #pragma unroll
      for (int r = 0; r < 8; ++r)
        sS[wv][(r + 8*hf) * 32 + 16*kn + nl] = s[r];
    }
    __syncthreads();
    if (lane < 16) { // online softmax bookkeeping, one row per lane
      int r = lane;
      float mOld = sM[wv][r], mNew = mOld;
      int kmax = SEQ - kbk * 32; if (kmax > 32) kmax = 32;
      for (int c = 0; c < kmax; ++c)
        mNew = fmaxf(mNew, sS[wv][r * 32 + c] * scale);
      float fac = __expf(mOld - mNew);
      float l = sL[wv][r] * fac;
      for (int c = 0; c < 32; ++c) {
        float p = 0.f;
        if (c < kmax) { p = __expf(sS[wv][r * 32 + c] * scale - mNew); l += p; }
        sP[wv][r * 32 + c] = (bf16_t)p;
      }
      sM[wv][r] = mNew; sL[wv][r] = l; sF[wv][r] = fac;
    }
    __syncthreads();
    float fr[8];
    #pragma unroll
    for (int r = 0; r < 8; ++r) fr[r] = sF[wv][r + 8*hf];
    v16bf pf = load_a16x32(sP[wv], 32, lane);
    #pragma unroll
    for (int cn = 0; cn < 4; ++cn) {
      #pragma unroll
      for (int r = 0; r < 8; ++r) accO[cn][r] *= fr[r];
      v16bf vf = load_b32x16(sV + 16*cn*40, 40, lane);
      accO[cn] = wmma_bf(pf, vf, accO[cn]);
    }
  }
  __syncthreads();
  if (lane < 16) { float l = sL[wv][lane]; sF[wv][lane] = (l > 0.f) ? 1.0f / l : 0.f; }
  __syncthreads();
  float inv[8];
  #pragma unroll
  for (int r = 0; r < 8; ++r) inv[r] = sF[wv][r + 8*hf];
  #pragma unroll
  for (int cn = 0; cn < 4; ++cn)
    #pragma unroll
    for (int r = 0; r < 8; ++r) {
      int row = q0 + r + 8*hf;
      if (row < SEQ)
        out[((size_t)b * SEQ + row) * CH + h * DH + 16*cn + nl] = (bf16_t)(accO[cn][r] * inv[r]);
    }
}

extern "C" void kernel_launch(void* const* d_in, const int* in_sizes, int n_in,
                              void* d_out, int out_size, void* d_ws, size_t ws_size,
                              hipStream_t stream) {
  const float* x     = (const float*)d_in[0];
  const float* sinp  = (const float*)d_in[1];
  const float* cosp  = (const float*)d_in[2];
  const float* Wqkv  = (const float*)d_in[3];
  const float* Wproj = (const float*)d_in[4];
  const float* bproj = (const float*)d_in[5];
  float* out = (float*)d_out;

  const size_t nx   = (size_t)BATCH * SEQ * CH;                 // 12,595,200
  const size_t nwq  = (size_t)CH * QKVN;
  const size_t nwp  = (size_t)CH * CH;
  const size_t nhkv = (size_t)BATCH * NUMH * SEQ * DH;          // per q/k/v

  char* ws = (char*)d_ws;
  bf16_t* xbf   = (bf16_t*)ws;                 ws += nx   * sizeof(bf16_t);  // reused as attn out
  bf16_t* wqT   = (bf16_t*)ws;                 ws += nwq  * sizeof(bf16_t);  // [2304][768]
  bf16_t* wpT   = (bf16_t*)ws;                 ws += nwp  * sizeof(bf16_t);  // [768][768]
  bf16_t* qkvbf = (bf16_t*)ws;                 ws += 3 * nhkv * sizeof(bf16_t);

  f32_to_bf16_kernel<<<(unsigned)((nx + 255) / 256), 256, 0, stream>>>(x, xbf, (int)nx);
  transpose_w_kernel<<<dim3(CH / 32, QKVN / 32), 256, 0, stream>>>(Wqkv,  wqT, CH, QKVN);
  transpose_w_kernel<<<dim3(CH / 32, CH / 32),   256, 0, stream>>>(Wproj, wpT, CH, CH);

  const int M = BATCH * SEQ;                   // 16400
  dim3 gq((M + 127) / 128, QKVN / 64);         // 129 x 36
  gemm_kernel<QKVN, 0><<<gq, 256, 0, stream>>>(xbf, wqT, qkvbf, nullptr, nullptr);

  const int nrope = 2 * BATCH * NUMH * 1024 * 32;  // 12,582,912
  rope_kernel<<<nrope / 256, 256, 0, stream>>>(qkvbf, sinp, cosp);

  dim3 ga(BATCH * NUMH, (SEQ + 127) / 128);    // 192 x 9
  flash_attn_kernel<<<ga, 256, 0, stream>>>(qkvbf, qkvbf + nhkv, qkvbf + 2 * nhkv, xbf);

  dim3 gp((M + 127) / 128, CH / 64);           // 129 x 12
  gemm_kernel<CH, 1><<<gp, 256, 0, stream>>>(xbf, wpT, nullptr, out, bproj);

  (void)in_sizes; (void)n_in; (void)out_size; (void)ws_size;
}
